// MultiHeadAttention_69243462746293
// MI455X (gfx1250) — compile-verified, matrix operand order checked by reading, not run on hardware
//
#include <hip/hip_runtime.h>
#include <hip/hip_bf16.h>

// ---------------------------------------------------------------------------
// MI455X (gfx1250) fused causal MHA forward.
// bf16 WMMA (v_wmma_f32_16x16x32_bf16) for all three matmuls; flash-style
// online-softmax attention (32 q-rows per wave for 2x K/V fragment reuse),
// P staged through per-wave LDS.
// ---------------------------------------------------------------------------

typedef __attribute__((ext_vector_type(16))) __bf16 v16bf;
typedef __attribute__((ext_vector_type(8)))  __bf16 v8bf;
typedef __attribute__((ext_vector_type(8)))  float  v8f;

#define B_  4
#define S_  2048
#define D_  1024
#define H_  16
#define HD_ 64

// ---- bf16 helpers (buffers stored as ushort, reinterpreted for fragments) --
__device__ __forceinline__ unsigned short f2bf(float f) {
  union { float f; unsigned int u; } c; c.f = f;
  unsigned int u = c.u;
  unsigned int r = u + 0x7FFFu + ((u >> 16) & 1u);     // round-to-nearest-even
  if ((u & 0x7F800000u) == 0x7F800000u) r = u;          // Inf/NaN: truncate
  return (unsigned short)(r >> 16);
}
__device__ __forceinline__ float bf2f(unsigned short h) {
  union { unsigned int u; float f; } c; c.u = ((unsigned int)h) << 16;
  return c.f;
}

// ---- 16x32 bf16 WMMA operand fragment load (A and B share this layout) ----
// lane<16 : row = lane,    elems = K[0..7], K[16..23]
// lane>=16: row = lane-16, elems = K[8..15], K[24..31]
__device__ __forceinline__ v16bf load_frag(const unsigned short* p, int ld) {
  const int lane = threadIdx.x & 31;
  const int row  = lane & 15;
  const int kb   = (lane >> 4) << 3;
  const unsigned short* q = p + row * ld + kb;
  union { v16bf v; v8bf h[2]; } u;
  u.h[0] = *(const v8bf*)(q);        // 16B load
  u.h[1] = *(const v8bf*)(q + 16);   // 16B load
  return u.v;
}

// ---------------------------------------------------------------------------
// f32 -> bf16 conversion
// ---------------------------------------------------------------------------
__global__ __launch_bounds__(256)
void cvt_bf16(const float* __restrict__ in, unsigned short* __restrict__ out, int n) {
  int i = blockIdx.x * blockDim.x + threadIdx.x;
  if (i < n) out[i] = f2bf(in[i]);
}

// ---------------------------------------------------------------------------
// QKV projection GEMM: C[m,c] = sum_k X[m,k] * Wqkv[c,k]
// Block tile 128x128 (4 waves, 64x64 per wave). Output scattered directly
// into Q[B,H,S,64], K[B,H,S,64], Vt[B,H,64,S] (bf16).
// ---------------------------------------------------------------------------
__global__ __launch_bounds__(128)
void gemm_qkv(const unsigned short* __restrict__ A,   // [8192,1024] bf16
              const unsigned short* __restrict__ W,   // [3072,1024] bf16
              unsigned short* __restrict__ qb,
              unsigned short* __restrict__ kb,
              unsigned short* __restrict__ vtb) {
  const int K  = D_;
  const int w  = threadIdx.x >> 5;
  const int m0 = blockIdx.x * 128 + (w >> 1) * 64;
  const int n0 = blockIdx.y * 128 + (w & 1) * 64;

  v8f z = {};
  v8f acc[4][4];
#pragma unroll
  for (int i = 0; i < 4; ++i)
#pragma unroll
    for (int j = 0; j < 4; ++j) acc[i][j] = z;

  for (int k0 = 0; k0 < K; k0 += 32) {
    v16bf af[4], bf[4];
#pragma unroll
    for (int i = 0; i < 4; ++i) af[i] = load_frag(A + (size_t)(m0 + 16 * i) * K + k0, K);
#pragma unroll
    for (int j = 0; j < 4; ++j) bf[j] = load_frag(W + (size_t)(n0 + 16 * j) * K + k0, K);
#pragma unroll
    for (int i = 0; i < 4; ++i)
#pragma unroll
      for (int j = 0; j < 4; ++j)
        acc[i][j] = __builtin_amdgcn_wmma_f32_16x16x32_bf16(
            false, af[i], false, bf[j], (short)0, acc[i][j], false, false);
  }

  const int lane = threadIdx.x & 31;
  const int n    = lane & 15;
  const int mb   = (lane >> 4) << 3;
#pragma unroll
  for (int i = 0; i < 4; ++i)
#pragma unroll
    for (int j = 0; j < 4; ++j)
#pragma unroll
      for (int r = 0; r < 8; ++r) {
        int m = m0 + 16 * i + mb + r;
        int c = n0 + 16 * j + n;
        int b   = m >> 11, s = m & (S_ - 1);
        int sec = c >> 10;
        int cc  = c & (D_ - 1);
        int h   = cc >> 6, d = cc & 63;
        int bh  = b * H_ + h;
        unsigned short bv = f2bf(acc[i][j][r]);
        if (sec == 0)      qb[((size_t)(bh * S_ + s) << 6) + d] = bv;
        else if (sec == 1) kb[((size_t)(bh * S_ + s) << 6) + d] = bv;
        else               vtb[(size_t)(bh * 64 + d) * S_ + s] = bv;
      }
}

// ---------------------------------------------------------------------------
// In-place RoPE on Q and K (head_dim 64, half = 32).
// one thread per (b,h,s,i), i in [0,32)
// ---------------------------------------------------------------------------
__global__ __launch_bounds__(256)
void rope_qk(unsigned short* __restrict__ qb, unsigned short* __restrict__ kb) {
  int idx = blockIdx.x * blockDim.x + threadIdx.x;
  int i  = idx & 31;
  int s  = (idx >> 5) & (S_ - 1);
  int bh = idx >> 16;                       // 32*2048 = 2^16 per (b,h)
  if (bh >= B_ * H_) return;
  size_t base = ((size_t)(bh * S_ + s)) << 6;
  float ang = (float)s * __powf(10000.0f, -(float)i * (1.0f / 32.0f));
  float sn, cs;
  __sincosf(ang, &sn, &cs);
  {
    float lo = bf2f(qb[base + i]), hi = bf2f(qb[base + 32 + i]);
    qb[base + i]      = f2bf(lo * cs - hi * sn);
    qb[base + 32 + i] = f2bf(lo * sn + hi * cs);
  }
  {
    float lo = bf2f(kb[base + i]), hi = bf2f(kb[base + 32 + i]);
    kb[base + i]      = f2bf(lo * cs - hi * sn);
    kb[base + 32 + i] = f2bf(lo * sn + hi * cs);
  }
}

// ---------------------------------------------------------------------------
// Causal flash attention. Grid: (B*H, S/256). 8 waves/block, 32 q-rows/wave
// (two 16-row tiles -> 2x reuse of every K and V fragment, 32 flops/byte).
// Per 32-key tile: 8 WMMAs for S = Q*K^T, online softmax in the C-layout
// (row == accumulator VGPR index, 16-lane shfl_xor row reductions), P staged
// through per-wave LDS (ds stores + s_wait_dscnt + ds b128 loads) into
// A-fragment layout, then 8 WMMAs for O += P*Vt.
// ---------------------------------------------------------------------------
__global__ __launch_bounds__(256)
void attn_fwd(const unsigned short* __restrict__ qb,
              const unsigned short* __restrict__ kb,
              const unsigned short* __restrict__ vtb,
              unsigned short* __restrict__ att) {
  __shared__ unsigned short plds[8][2 * 16 * 32];   // 2KB per wave

  const int bh   = blockIdx.x;
  const int b    = bh >> 4, h = bh & 15;
  const int w    = threadIdx.x >> 5;
  const int lane = threadIdx.x & 31;
  const int q0   = blockIdx.y * 256 + w * 32;       // first of 32 rows
  const int n    = lane & 15;
  const int mb   = (lane >> 4) << 3;

  const unsigned short* qp = qb  + ((size_t)(bh * S_ + q0) << 6);
  const unsigned short* kp = kb  + ((size_t)(bh * S_) << 6);
  const unsigned short* vp = vtb + (size_t)(bh * 64) * S_;

  // Q fragments: [row-tile][k-half]
  v16bf aq[2][2];
#pragma unroll
  for (int ri = 0; ri < 2; ++ri) {
    aq[ri][0] = load_frag(qp + (size_t)(16 * ri) * 64, 64);
    aq[ri][1] = load_frag(qp + (size_t)(16 * ri) * 64 + 32, 64);
  }

  v8f z = {};
  v8f o[2][4];
#pragma unroll
  for (int ri = 0; ri < 2; ++ri)
#pragma unroll
    for (int d = 0; d < 4; ++d) o[ri][d] = z;
  float mrun[2][8], lrun[2][8];
#pragma unroll
  for (int ri = 0; ri < 2; ++ri)
#pragma unroll
    for (int r = 0; r < 8; ++r) { mrun[ri][r] = -3.0e38f; lrun[ri][r] = 0.0f; }

  const float sc = 0.125f;              // 1/sqrt(64)
  const int kend = q0 + 32;             // causal bound (exclusive), per wave
  unsigned short* myp = plds[w];

  for (int j = 0; j < kend; j += 32) {
    // ---- K fragments, shared across both row tiles ----
    v16bf bk[2][2];                     // [key-subtile][k-half]
#pragma unroll
    for (int t = 0; t < 2; ++t) {
      bk[t][0] = load_frag(kp + (size_t)(j + 16 * t) * 64, 64);
      bk[t][1] = load_frag(kp + (size_t)(j + 16 * t) * 64 + 32, 64);
    }
    // ---- scores: 2 row tiles x 2 key subtiles ----
    v8f st[2][2];
#pragma unroll
    for (int ri = 0; ri < 2; ++ri)
#pragma unroll
      for (int t = 0; t < 2; ++t) {
        v8f s0 = z;
        s0 = __builtin_amdgcn_wmma_f32_16x16x32_bf16(false, aq[ri][0], false, bk[t][0], (short)0, s0, false, false);
        s0 = __builtin_amdgcn_wmma_f32_16x16x32_bf16(false, aq[ri][1], false, bk[t][1], (short)0, s0, false, false);
        st[ri][t] = s0;
      }
    // ---- per row tile: scale + causal mask + online softmax + stage P ----
#pragma unroll
    for (int ri = 0; ri < 2; ++ri) {
#pragma unroll
      for (int r = 0; r < 8; ++r) {
        int row = q0 + 16 * ri + mb + r;
        float s0 = st[ri][0][r] * sc;
        float s1 = st[ri][1][r] * sc;
        if (j + n > row)      s0 = -3.0e38f;
        if (j + 16 + n > row) s1 = -3.0e38f;
        float mx = fmaxf(s0, s1);
#pragma unroll
        for (int off = 1; off < 16; off <<= 1)
          mx = fmaxf(mx, __shfl_xor(mx, off, 32));
        float mnew = fmaxf(mrun[ri][r], mx);
        float p0 = __expf(s0 - mnew);
        float p1 = __expf(s1 - mnew);
        float rs = p0 + p1;
#pragma unroll
        for (int off = 1; off < 16; off <<= 1)
          rs += __shfl_xor(rs, off, 32);
        float corr = __expf(mrun[ri][r] - mnew);
        lrun[ri][r] = lrun[ri][r] * corr + rs;
        mrun[ri][r] = mnew;
#pragma unroll
        for (int d = 0; d < 4; ++d) o[ri][d][r] = o[ri][d][r] * corr;
        myp[ri * 512 + (mb + r) * 32 + n]      = f2bf(p0);
        myp[ri * 512 + (mb + r) * 32 + 16 + n] = f2bf(p1);
      }
    }
    __builtin_amdgcn_wave_barrier();
    asm volatile("s_wait_dscnt 0x0" ::: "memory");
    // ---- O += P @ Vt : V fragments shared across both row tiles ----
    v16bf pa[2];
    pa[0] = load_frag(myp, 32);
    pa[1] = load_frag(myp + 512, 32);
#pragma unroll
    for (int d = 0; d < 4; ++d) {
      v16bf bv = load_frag(vp + (size_t)(16 * d) * S_ + j, S_);
#pragma unroll
      for (int ri = 0; ri < 2; ++ri)
        o[ri][d] = __builtin_amdgcn_wmma_f32_16x16x32_bf16(false, pa[ri], false, bv, (short)0, o[ri][d], false, false);
    }
  }

  // ---- normalize and emit bf16 in [B*S, D] layout for the out projection ----
#pragma unroll
  for (int ri = 0; ri < 2; ++ri) {
    unsigned short* op = att + ((size_t)(b * S_ + q0 + 16 * ri) << 10) + h * 64;
#pragma unroll
    for (int r = 0; r < 8; ++r) {
      float inv = 1.0f / lrun[ri][r];
#pragma unroll
      for (int d = 0; d < 4; ++d)
        op[(size_t)(mb + r) * D_ + 16 * d + n] = f2bf(o[ri][d][r] * inv);
    }
  }
}

// ---------------------------------------------------------------------------
// Output projection GEMM: out[m,c] = sum_k Attn[m,k] * Wout[c,k], f32 out.
// ---------------------------------------------------------------------------
__global__ __launch_bounds__(128)
void gemm_out(const unsigned short* __restrict__ A,   // [8192,1024] bf16
              const unsigned short* __restrict__ W,   // [1024,1024] bf16
              float* __restrict__ C) {
  const int K  = D_, N = D_;
  const int w  = threadIdx.x >> 5;
  const int m0 = blockIdx.x * 128 + (w >> 1) * 64;
  const int n0 = blockIdx.y * 128 + (w & 1) * 64;

  v8f z = {};
  v8f acc[4][4];
#pragma unroll
  for (int i = 0; i < 4; ++i)
#pragma unroll
    for (int j = 0; j < 4; ++j) acc[i][j] = z;

  for (int k0 = 0; k0 < K; k0 += 32) {
    v16bf af[4], bf[4];
#pragma unroll
    for (int i = 0; i < 4; ++i) af[i] = load_frag(A + (size_t)(m0 + 16 * i) * K + k0, K);
#pragma unroll
    for (int j = 0; j < 4; ++j) bf[j] = load_frag(W + (size_t)(n0 + 16 * j) * K + k0, K);
#pragma unroll
    for (int i = 0; i < 4; ++i)
#pragma unroll
      for (int j = 0; j < 4; ++j)
        acc[i][j] = __builtin_amdgcn_wmma_f32_16x16x32_bf16(
            false, af[i], false, bf[j], (short)0, acc[i][j], false, false);
  }

  const int lane = threadIdx.x & 31;
  const int n    = lane & 15;
  const int mb   = (lane >> 4) << 3;
#pragma unroll
  for (int i = 0; i < 4; ++i)
#pragma unroll
    for (int j = 0; j < 4; ++j)
#pragma unroll
      for (int r = 0; r < 8; ++r)
        C[(size_t)(m0 + 16 * i + mb + r) * N + (n0 + 16 * j + n)] = acc[i][j][r];
}

// ---------------------------------------------------------------------------
extern "C" void kernel_launch(void* const* d_in, const int* in_sizes, int n_in,
                              void* d_out, int out_size, void* d_ws, size_t ws_size,
                              hipStream_t stream) {
  (void)in_sizes; (void)n_in; (void)out_size; (void)ws_size;
  const float* x    = (const float*)d_in[0];
  const float* wqkv = (const float*)d_in[1];
  const float* wout = (const float*)d_in[2];
  float* out = (float*)d_out;

  const size_t NX  = (size_t)B_ * S_ * D_;   // 8,388,608
  const size_t NWQ = (size_t)3 * D_ * D_;    // 3,145,728
  const size_t NWO = (size_t)D_ * D_;        // 1,048,576

  unsigned short* ws    = (unsigned short*)d_ws;
  unsigned short* xb    = ws;          ws += NX;
  unsigned short* wqkvb = ws;          ws += NWQ;
  unsigned short* woutb = ws;          ws += NWO;
  unsigned short* qb    = ws;          ws += NX;
  unsigned short* kb    = ws;          ws += NX;
  unsigned short* vtb   = ws;          ws += NX;
  unsigned short* att   = ws;          // +NX  (total ~92 MB)

  cvt_bf16<<<(unsigned)((NX  + 255) / 256), 256, 0, stream>>>(x,    xb,    (int)NX);
  cvt_bf16<<<(unsigned)((NWQ + 255) / 256), 256, 0, stream>>>(wqkv, wqkvb, (int)NWQ);
  cvt_bf16<<<(unsigned)((NWO + 255) / 256), 256, 0, stream>>>(wout, woutb, (int)NWO);

  gemm_qkv<<<dim3(8192 / 128, 3072 / 128), 128, 0, stream>>>(xb, wqkvb, qb, kb, vtb);
  rope_qk<<<(B_ * H_ * S_ * 32) / 256, 256, 0, stream>>>(qb, kb);
  attn_fwd<<<dim3(B_ * H_, S_ / 256), 256, 0, stream>>>(qb, kb, vtb, att);
  gemm_out<<<dim3(8192 / 128, 1024 / 128), 128, 0, stream>>>(att, woutb, out);
}
